// Taco2ProdDecoderCell_11355893531396
// MI455X (gfx1250) — matrix operand order repaired and checked by reading, not compiled, not run on hardware
//
#include <hip/hip_runtime.h>
#include <hip/hip_bf16.h>

// ---------------------------------------------------------------------------
// Taco2 decoder cell on gfx1250 (CDNA5). GEMMs via v_wmma_f32_16x16x32_bf16.
// memory tensor staged once per tile with global_load_async_to_lds_b128 and
// consumed by BOTH the energy WMMA and the context reduction (fused), with a
// boundary-energy WMMA pre-pass supplying the monotonic-attention carry-in.
// ---------------------------------------------------------------------------

typedef __attribute__((ext_vector_type(16))) __bf16 v16bf;
typedef __attribute__((ext_vector_type(8)))  float  v8f;

union BF16x16 {
    v16bf v;
    __bf16 h[16];
    uint4  q4[2];
};

__device__ __forceinline__ void pack4(__bf16* d, float4 f) {
    d[0] = (__bf16)f.x; d[1] = (__bf16)f.y; d[2] = (__bf16)f.z; d[3] = (__bf16)f.w;
}

__device__ __forceinline__ float sigmoidf_(float x) {
    return 1.0f / (1.0f + __expf(-x));
}

// One K=32 WMMA chunk from fp32 sources (used by the LSTM-gate GEMMs).
__device__ __forceinline__ v8f gchunk(const float* __restrict__ arow,
                                      const float* __restrict__ wrow,
                                      int abase, int bbase, v8f acc) {
    const float4* ap = (const float4*)(arow + abase);
    const float4* aq = (const float4*)(arow + abase + 16);
    const float4* bp = (const float4*)(wrow + bbase);
    const float4* bq = (const float4*)(wrow + bbase + 16);
    float4 a0 = ap[0], a1 = ap[1], a2 = aq[0], a3 = aq[1];
    float4 b0 = bp[0], b1 = bp[1], b2 = bq[0], b3 = bq[1];
    BF16x16 A, B;
    pack4(A.h + 0, a0); pack4(A.h + 4, a1); pack4(A.h + 8, a2); pack4(A.h + 12, a3);
    pack4(B.h + 0, b0); pack4(B.h + 4, b1); pack4(B.h + 8, b2); pack4(B.h + 12, b3);
    return __builtin_amdgcn_wmma_f32_16x16x32_bf16(
        false, A.v, false, B.v, (short)0, acc, false, false);
}

// ---------------------------------------------------------------------------
// PreNet (deterministic hash dropout, p=0.5, scale 2).
// ---------------------------------------------------------------------------
__global__ void prenet_kernel(const float* __restrict__ x,
                              const float* __restrict__ W1, const float* __restrict__ b1,
                              const float* __restrict__ W2, const float* __restrict__ b2,
                              float* __restrict__ xpre) {
    const int b = blockIdx.x;       // 64
    const int p = threadIdx.x;      // 128
    __shared__ float xf[80];
    __shared__ float h1[128];
    if (p < 80) xf[p] = x[b * 80 + p];
    __syncthreads();

    float s = b1[p];
    #pragma unroll 4
    for (int k = 0; k < 80; ++k) s += xf[k] * W1[p * 80 + k];
    s = fmaxf(s, 0.0f);
    unsigned int hA = (unsigned int)(b * 128 + p) * 2654435761u + 0x9E3779B9u;
    hA ^= hA >> 16; hA *= 0x45d9f3bu; hA ^= hA >> 13;
    s = (hA & 1u) ? (2.0f * s) : 0.0f;
    h1[p] = s;
    __syncthreads();

    float t = b2[p];
    #pragma unroll 4
    for (int k = 0; k < 128; ++k) t += h1[k] * W2[p * 128 + k];
    t = fmaxf(t, 0.0f);
    unsigned int hB = (unsigned int)(b * 128 + p) * 2246822519u + 0x85EBCA6Bu;
    hB ^= hB >> 16; hB *= 0x45d9f3bu; hB ^= hB >> 13;
    t = (hB & 1u) ? (2.0f * t) : 0.0f;
    xpre[b * 128 + p] = t;
}

// ---------------------------------------------------------------------------
// LSTM gate GEMM:  G[64,Nout] = [src0|src1] @ Wih^T + h @ Whh^T + bih + bhh
// ---------------------------------------------------------------------------
__global__ void lstm_gates_wmma(const float* __restrict__ src0, int w0,
                                const float* __restrict__ src1, int w1,
                                const float* __restrict__ hsrc, int wh,
                                const float* __restrict__ Wih,
                                const float* __restrict__ Whh,
                                const float* __restrict__ bih,
                                const float* __restrict__ bhh,
                                float* __restrict__ gout, int Nout) {
    const int lane = threadIdx.x & 31;
    const int wave = threadIdx.x >> 5;
    const int tile = blockIdx.x * 8 + wave;
    const int ntn  = Nout >> 4;
    const int m0   = (tile / ntn) << 4;
    const int n0   = (tile % ntn) << 4;
    const int half = lane >> 4;
    const int l15  = lane & 15;

    const int Kih   = w0 + w1;
    const int am    = m0 + l15;
    const int bn    = n0 + l15;
    const int abase = half * 8;
    const int bbase = half * 8;

    const float* a0p = src0 + (size_t)am * w0;
    const float* a1p = src1 + (size_t)am * w1;
    const float* ahp = hsrc + (size_t)am * wh;
    const float* wih = Wih  + (size_t)bn * Kih;
    const float* whh = Whh  + (size_t)bn * wh;

    v8f acc = {};
    for (int k0 = 0; k0 < w0; k0 += 32) acc = gchunk(a0p + k0, wih + k0,      abase, bbase, acc);
    for (int k0 = 0; k0 < w1; k0 += 32) acc = gchunk(a1p + k0, wih + w0 + k0, abase, bbase, acc);
    for (int k0 = 0; k0 < wh; k0 += 32) acc = gchunk(ahp + k0, whh + k0,      abase, bbase, acc);

    const int n = n0 + l15;
    const float bias = bih[n] + bhh[n];
    #pragma unroll
    for (int r = 0; r < 8; ++r) {
        const int m = m0 + r + half * 8;
        gout[(size_t)m * Nout + n] = acc[r] + bias;
    }
}

// ---------------------------------------------------------------------------
// LSTM pointwise + zoneout.
// ---------------------------------------------------------------------------
__global__ void lstm_pw(const float* __restrict__ g,
                        const float* __restrict__ h_old, const float* __restrict__ c_old,
                        float* __restrict__ h_new, float* __restrict__ c_new, int H) {
    const int idx = blockIdx.x * blockDim.x + threadIdx.x;
    const int b = idx / H, j = idx - b * H;
    const float* gr = g + (size_t)b * 4 * H;
    const float i  = gr[j];
    const float f  = gr[H + j];
    const float gg = gr[2 * H + j];
    const float o  = gr[3 * H + j];
    const float cn = sigmoidf_(f) * c_old[idx] + sigmoidf_(i) * tanhf(gg);
    const float hn = sigmoidf_(o) * tanhf(cn);
    h_new[idx] = 0.9f * hn + 0.1f * h_old[idx];
    c_new[idx] = 0.9f * cn + 0.1f * c_old[idx];
}

// ---------------------------------------------------------------------------
// q[b,a] = ah[b] . Wq[a] + bq[a]
// ---------------------------------------------------------------------------
__global__ void q_kernel(const float* __restrict__ ah, const float* __restrict__ Wq,
                         const float* __restrict__ bq, float* __restrict__ q) {
    const int idx = blockIdx.x * blockDim.x + threadIdx.x;   // 8192
    const int b = idx >> 7, a = idx & 127;
    const float* hr = ah + (size_t)b * 1024;
    const float* wr = Wq + (size_t)a * 1024;
    float s = bq[a];
    #pragma unroll 8
    for (int k = 0; k < 1024; ++k) s += hr[k] * wr[k];
    q[idx] = s;
}

// ---------------------------------------------------------------------------
// att_Wm -> bf16 (once per call; 65536 elems)
// ---------------------------------------------------------------------------
__global__ void cvt_wm_kernel(const float* __restrict__ Wm, unsigned short* __restrict__ wmb) {
    const int i = blockIdx.x * blockDim.x + threadIdx.x;
    wmb[i] = __builtin_bit_cast(unsigned short, (__bf16)Wm[i]);
}

// ---------------------------------------------------------------------------
// Shared tile-energy engine: stages 16 rows (t = tbase + r*tstride) of memory
// into LDS (async b128, ASYNCcnt), converts once to a padded bf16 tile, runs
// 8 waves x 16 WMMAs against bf16 Wm, then reduces e per row with a
// fixed-order shfl_xor tree (deterministic). Leaves fp32 tile in lds_f.
// ---------------------------------------------------------------------------
__device__ __forceinline__ void tile_energies(
    const float* __restrict__ mem, const unsigned short* __restrict__ wmb,
    const float* __restrict__ bm, const float* __restrict__ q,
    const float* __restrict__ vvec,
    int b, int tbase, int tstride,
    float* lds_f, unsigned short* lds_a, float* ewave, float* esh) {

    const int lane = threadIdx.x & 31;
    const int wave = threadIdx.x >> 5;          // n-tile 0..7
    const int n0   = wave << 4;
    const int half = lane >> 4;
    const int l15  = lane & 15;

    // async stage: 2048 x 16B transfers, 8 per thread
    {
        const unsigned lds_base = (unsigned)(uintptr_t)lds_f;
        for (int i = threadIdx.x; i < 2048; i += 256) {
            const int row  = i >> 7;            // 128 float4 per row
            const int col4 = i & 127;
            const unsigned dst = lds_base + (unsigned)i * 16u;
            const float* g = mem + ((size_t)(b * 2048 + tbase + row * tstride)) * 512
                                 + (size_t)col4 * 4;
            asm volatile("global_load_async_to_lds_b128 %0, %1, off"
                         :: "v"(dst), "v"(g) : "memory");
        }
        asm volatile("s_wait_asynccnt 0" ::: "memory");
    }
    __syncthreads();

    // one-shot fp32 -> bf16 into padded tile
    for (int i = threadIdx.x; i < 16 * 512; i += 256) {
        const int row = i >> 9, k = i & 511;
        lds_a[row * 520 + k] = __builtin_bit_cast(unsigned short, (__bf16)lds_f[i]);
    }
    __syncthreads();

    const unsigned short* arow = lds_a + l15 * 520;
    const unsigned short* wrow = wmb + (size_t)(n0 + l15) * 512;
    const int abase = half * 8;
    const int bbase = half * 8;

    v8f acc = {};
    for (int k0 = 0; k0 < 512; k0 += 32) {
        BF16x16 A, B;
        A.q4[0] = *(const uint4*)(arow + k0 + abase);        // ds_load_b128
        A.q4[1] = *(const uint4*)(arow + k0 + abase + 16);
        B.q4[0] = *(const uint4*)(wrow + k0 + bbase);        // bf16 weights, no cvt
        B.q4[1] = *(const uint4*)(wrow + k0 + bbase + 16);
        acc = __builtin_amdgcn_wmma_f32_16x16x32_bf16(
            false, A.v, false, B.v, (short)0, acc, false, false);
    }

    const int n = n0 + l15;
    const float qv = q[b * 128 + n] + bm[n];
    const float vv = vvec[n];
    #pragma unroll
    for (int r = 0; r < 8; ++r) {
        float v = tanhf(acc[r] + qv) * vv;
        v += __shfl_xor(v, 1, 32);               // fixed-order 16-lane tree
        v += __shfl_xor(v, 2, 32);
        v += __shfl_xor(v, 4, 32);
        v += __shfl_xor(v, 8, 32);
        if (l15 == 0) ewave[wave * 16 + half * 8 + r] = v;
    }
    __syncthreads();
    if (threadIdx.x < 16) {
        float s = 0.0f;
        #pragma unroll
        for (int w = 0; w < 8; ++w) s += ewave[w * 16 + threadIdx.x];
        esh[threadIdx.x] = s;
    }
    __syncthreads();
}

// ---------------------------------------------------------------------------
// Pre-pass: energies of boundary rows t = 16j+15 only (carry-in for fusion).
// Grid (8, 64).
// ---------------------------------------------------------------------------
__global__ void energy_boundary(const float* __restrict__ mem,
                                const unsigned short* __restrict__ wmb,
                                const float* __restrict__ bm,
                                const float* __restrict__ q,
                                const float* __restrict__ vvec,
                                float* __restrict__ eb) {        // [64,128]
    __shared__ float          lds_f[16 * 512];
    __shared__ unsigned short lds_a[16 * 520];
    __shared__ float          ewave[128];
    __shared__ float          esh[16];
    const int b  = blockIdx.y;
    const int j0 = blockIdx.x << 4;
    tile_energies(mem, wmb, bm, q, vvec, b, j0 * 16 + 15, 16, lds_f, lds_a, ewave, esh);
    if (threadIdx.x < 16) eb[b * 128 + j0 + threadIdx.x] = esh[threadIdx.x];
}

// ---------------------------------------------------------------------------
// Main pass. mode 0 (fused): energies -> w_new rows + context partial from the
// already-staged fp32 tile (memory read ONCE).  mode 1 (fallback): write e.
// Grid (128, 64).
// ---------------------------------------------------------------------------
__global__ void energy_fused(const float* __restrict__ mem,
                             const unsigned short* __restrict__ wmb,
                             const float* __restrict__ bm,
                             const float* __restrict__ q,
                             const float* __restrict__ vvec,
                             const float* __restrict__ watt,
                             const unsigned char* __restrict__ mmask,
                             const float* __restrict__ eb,       // [64,128]
                             float* __restrict__ wnew,           // [64,2048]
                             float* __restrict__ ctxpart,        // [64,128,512]
                             float* __restrict__ e_out,          // [64,2048]
                             int mode) {
    __shared__ float          lds_f[16 * 512];
    __shared__ unsigned short lds_a[16 * 520];
    __shared__ float          ewave[128];
    __shared__ float          esh[16];
    __shared__ float          wnsh[16];

    const int b    = blockIdx.y;
    const int tblk = blockIdx.x;
    const int t0   = tblk << 4;
    tile_energies(mem, wmb, bm, q, vvec, b, t0, 1, lds_f, lds_a, ewave, esh);

    if (mode == 1) {
        if (threadIdx.x < 16) e_out[b * 2048 + t0 + threadIdx.x] = esh[threadIdx.x];
        return;
    }

    if (threadIdx.x < 16) {
        const int r = threadIdx.x;
        const int t = t0 + r;
        const float p = sigmoidf_(esh[r]);
        float prevterm;
        if (r > 0)          prevterm = watt[b * 2048 + t - 1] * (1.0f - sigmoidf_(esh[r - 1]));
        else if (tblk > 0)  prevterm = watt[b * 2048 + t - 1] * (1.0f - sigmoidf_(eb[b * 128 + tblk - 1]));
        else                prevterm = 0.0f;
        float v = watt[b * 2048 + t] * p + prevterm;
        v = mmask[b * 2048 + t] ? v : 0.0f;
        wnsh[r] = v;
        wnew[b * 2048 + t] = v;
    }
    __syncthreads();

    // context partial from the fp32 tile already in LDS
    #pragma unroll
    for (int cc = 0; cc < 2; ++cc) {
        const int c = threadIdx.x + (cc << 8);
        float s = 0.0f;
        #pragma unroll
        for (int r = 0; r < 16; ++r) s += wnsh[r] * lds_f[r * 512 + c];
        ctxpart[(((size_t)b * 128 + tblk) << 9) + c] = s;
    }
}

// ---------------------------------------------------------------------------
// ctx[b,c] = sum over 128 tile partials (fixed order, deterministic).
// ---------------------------------------------------------------------------
__global__ void ctx_reduce(const float* __restrict__ ctxpart, float* __restrict__ ctx) {
    const int idx = blockIdx.x * blockDim.x + threadIdx.x;   // 32768
    const int b = idx >> 9, c = idx & 511;
    const float* p = ctxpart + ((size_t)b << 16) + c;
    float s = 0.0f;
    #pragma unroll 8
    for (int j = 0; j < 128; ++j) s += p[(size_t)j << 9];
    ctx[idx] = s;
}

// --------------------- fallback path (small workspace) ---------------------
__global__ void wnew_kernel(const float* __restrict__ e, const float* __restrict__ watt,
                            const unsigned char* __restrict__ mmask,
                            float* __restrict__ wnew) {
    const int idx = blockIdx.x * blockDim.x + threadIdx.x;   // 131072
    const int t = idx & 2047;
    float v = watt[idx] * sigmoidf_(e[idx]);
    if (t > 0) v += watt[idx - 1] * (1.0f - sigmoidf_(e[idx - 1]));
    wnew[idx] = mmask[idx] ? v : 0.0f;
}

__global__ void ctx_kernel(const float* __restrict__ mem, const float* __restrict__ wnew,
                           float* __restrict__ ctx) {
    const int b  = blockIdx.y;
    const int c0 = blockIdx.x * 64;
    const int cc = threadIdx.x & 63;
    const int tt = threadIdx.x >> 6;
    const float* mb = mem + (size_t)b * 2048 * 512;
    const float* wb = wnew + b * 2048;
    float s = 0.0f;
    for (int t = tt; t < 2048; t += 4) {
        __builtin_prefetch(&mb[(size_t)(t + 8) * 512 + c0 + cc], 0, 1);
        s += wb[t] * mb[(size_t)t * 512 + c0 + cc];
    }
    __shared__ float red[256];
    red[threadIdx.x] = s;
    __syncthreads();
    if (tt == 0)
        ctx[b * 512 + c0 + cc] = red[cc] + red[cc + 64] + red[cc + 128] + red[cc + 192];
}

// ---------------------------------------------------------------------------
// x_dec = [dh | ctx]
// ---------------------------------------------------------------------------
__global__ void xdec_kernel(const float* __restrict__ dh, const float* __restrict__ ctx,
                            float* __restrict__ xdec) {
    const int idx = blockIdx.x * blockDim.x + threadIdx.x;   // 98304
    const int b = idx / 1536, j = idx - b * 1536;
    xdec[idx] = (j < 1024) ? dh[b * 1024 + j] : ctx[b * 512 + (j - 1024)];
}

// ---------------------------------------------------------------------------
extern "C" void kernel_launch(void* const* d_in, const int* in_sizes, int n_in,
                              void* d_out, int out_size, void* d_ws, size_t ws_size,
                              hipStream_t stream) {
    const float* x        = (const float*)d_in[0];
    const float* w_att    = (const float*)d_in[1];
    const float* ctx_att  = (const float*)d_in[2];
    const float* h_att_h  = (const float*)d_in[3];
    const float* h_att_c  = (const float*)d_in[4];
    const float* h_dec_h  = (const float*)d_in[5];
    const float* h_dec_c  = (const float*)d_in[6];
    const float* memory   = (const float*)d_in[7];
    const unsigned char* mmask = (const unsigned char*)d_in[8];
    const float* pre_W1   = (const float*)d_in[9];
    const float* pre_b1   = (const float*)d_in[10];
    const float* pre_W2   = (const float*)d_in[11];
    const float* pre_b2   = (const float*)d_in[12];
    const float* att_Wq   = (const float*)d_in[13];
    const float* att_bq   = (const float*)d_in[14];
    const float* att_Wm   = (const float*)d_in[15];
    const float* att_bm   = (const float*)d_in[16];
    const float* att_v    = (const float*)d_in[17];
    const float* arn_Wih  = (const float*)d_in[18];
    const float* arn_Whh  = (const float*)d_in[19];
    const float* arn_bih  = (const float*)d_in[20];
    const float* arn_bhh  = (const float*)d_in[21];
    const float* drn_Wih  = (const float*)d_in[22];
    const float* drn_Whh  = (const float*)d_in[23];
    const float* drn_bih  = (const float*)d_in[24];
    const float* drn_bhh  = (const float*)d_in[25];

    // Output layout (flat, return order): x_dec, ctx, w_new, ah, ac, dh, dc
    float* out  = (float*)d_out;
    float* xdec = out;                 // 64*1536
    float* ctx  = out + 98304;         // 64*512
    float* wnew = out + 131072;        // 64*2048
    float* ah   = out + 262144;
    float* ac   = out + 327680;
    float* dh   = out + 393216;
    float* dc   = out + 458752;

    // Workspace layout (float units)
    float* W = (float*)d_ws;
    float*          xpre    = W;                        // [0,      8192)
    float*          q       = W + 8192;                 // [8192,  16384)
    float*          eb      = W + 16384;                // [16384, 24576)
    unsigned short* wmb     = (unsigned short*)(W + 24576);  // 65536 u16 -> [24576,57344)
    float*          e       = W + 57344;                // [57344, 188416)
    float*          gates   = W + 188416;               // [188416,450560)
    float*          ctxpart = W + 450560;               // [450560,4644864)
    const bool fused = ws_size >= (size_t)4644864 * sizeof(float);

    // 1. PreNet
    prenet_kernel<<<64, 128, 0, stream>>>(x, pre_W1, pre_b1, pre_W2, pre_b2, xpre);

    // 2-3. Attention LSTM
    lstm_gates_wmma<<<128, 256, 0, stream>>>(xpre, 128, ctx_att, 512, h_att_h, 1024,
                                             arn_Wih, arn_Whh, arn_bih, arn_bhh,
                                             gates, 4096);
    lstm_pw<<<256, 256, 0, stream>>>(gates, h_att_h, h_att_c, ah, ac, 1024);

    // 4. Query projection + bf16 Wm
    q_kernel<<<32, 256, 0, stream>>>(ah, att_Wq, att_bq, q);
    cvt_wm_kernel<<<256, 256, 0, stream>>>(att_Wm, wmb);

    // 5-7. Attention energies + w_new + context
    if (fused) {
        energy_boundary<<<dim3(8, 64), 256, 0, stream>>>(memory, wmb, att_bm, q, att_v, eb);
        energy_fused<<<dim3(128, 64), 256, 0, stream>>>(memory, wmb, att_bm, q, att_v,
                                                        w_att, mmask, eb, wnew, ctxpart,
                                                        e, 0);
        ctx_reduce<<<128, 256, 0, stream>>>(ctxpart, ctx);
    } else {
        energy_fused<<<dim3(128, 64), 256, 0, stream>>>(memory, wmb, att_bm, q, att_v,
                                                        w_att, mmask, eb, wnew, ctxpart,
                                                        e, 1);
        wnew_kernel<<<512, 256, 0, stream>>>(e, w_att, mmask, wnew);
        ctx_kernel<<<dim3(8, 64), 256, 0, stream>>>(memory, wnew, ctx);
    }

    // 8-9. Decoder LSTM
    lstm_gates_wmma<<<128, 256, 0, stream>>>(ah, 1024, ctx, 512, h_dec_h, 1024,
                                             drn_Wih, drn_Whh, drn_bih, drn_bhh,
                                             gates, 4096);
    lstm_pw<<<256, 256, 0, stream>>>(gates, h_dec_h, h_dec_c, dh, dc, 1024);

    // 10. x_dec = [dh | ctx]
    xdec_kernel<<<384, 256, 0, stream>>>(dh, ctx, xdec);
}
